// FlashMultiHeadAttention_57887569215842
// MI455X (gfx1250) — matrix operand order repaired and checked by reading, not compiled, not run on hardware
//
#include <hip/hip_runtime.h>

// ---------------------------------------------------------------------------
// GQA causal attention layer for MI455X (gfx1250).
// bf16 WMMA everywhere; GEMM B-panels staged to LDS by the Tensor Data Mover
// (tensor_load_to_lds, TENSORcnt-synchronized, double buffered).
// B=2, N=2048, E=2048, HQ=32, HKV=8, D=64.
// ---------------------------------------------------------------------------

typedef __attribute__((ext_vector_type(16))) __bf16 v16bf;
typedef __attribute__((ext_vector_type(8)))  __bf16 v8bf;
typedef __attribute__((ext_vector_type(8)))  float  v8f;
typedef __attribute__((ext_vector_type(4)))  unsigned int v4u;
typedef __attribute__((ext_vector_type(8)))  int v8i;
typedef __attribute__((ext_vector_type(4)))  int v4i;

#define B_   2
#define N_   2048
#define E_   2048
#define HQ_  32
#define HKV_ 8
#define D_   64
#define G_   (HQ_ / HKV_)       // 4 query heads per kv head
#define SCALE_ 0.125f           // 1/sqrt(64)

#define OUT_BF16_PLAIN 0
#define OUT_KT         1
#define OUT_V          2
#define OUT_F32        3

// LDS B-panel row stride: 64 cols * 2B = 128B data + 16B TDM pad = 144B (72 el)
#define BSTRIDE_EL 72

static __device__ __forceinline__ v8f wmma_bf16(v16bf a, v16bf b, v8f c) {
  return __builtin_amdgcn_wmma_f32_16x16x32_bf16(false, a, false, b,
                                                 (short)0, c, false, false);
}

// ---- Tensor Data Mover: 2D bf16 tile (rowlen x rows) global -> LDS --------
// Writes rows at a 144B stride via D# pad fields (interval 32 DW, amount 4 DW).
static __device__ __forceinline__ void tdm_load_2d_b(
    const __bf16* gptr, unsigned int lds_off,
    unsigned int tensor_d0, unsigned int tensor_d1,
    unsigned int rowlen_el, unsigned int rows, unsigned long long stride_el) {
  unsigned long long ga = (unsigned long long)(uintptr_t)gptr;
  v4u g0;
  g0[0] = 1u;                                               // count=1 (valid user D#)
  g0[1] = lds_off;                                          // lds_addr
  g0[2] = (unsigned int)ga;                                 // global_addr[31:0]
  g0[3] = ((unsigned int)(ga >> 32) & 0x01FFFFFFu)          // global_addr[56:32]
        | (2u << 30);                                       // type=2 ("image")
  v8i g1;
  g1[0] = (int)((1u << 16)        // data_size = 2 bytes
              | (1u << 20)        // pad_enable
              | (4u << 22)        // pad_interval: 32 DWORDs (=128B row)
              | (3u << 25));      // pad_amount:   4 DWORDs (=16B)
  g1[1] = (int)((tensor_d0 & 0xFFFFu) << 16);               // [47:32]=0, [63:48]=td0.lo
  g1[2] = (int)((tensor_d0 >> 16) | ((tensor_d1 & 0xFFFFu) << 16));
  g1[3] = (int)((tensor_d1 >> 16) | (rowlen_el << 16));     // tile_dim0
  g1[4] = (int)rows;                                        // tile_dim1 (tile_dim2=0)
  g1[5] = (int)(unsigned int)stride_el;                     // dim0_stride[31:0]
  g1[6] = (int)(unsigned int)((stride_el >> 32) & 0xFFFFu); // dim0_stride[47:32]
  g1[7] = 0;
  v4i gz = {0, 0, 0, 0};
#if __clang_major__ >= 23
  v8i gz8 = {0, 0, 0, 0, 0, 0, 0, 0};
  __builtin_amdgcn_tensor_load_to_lds(g0, g1, gz, gz, gz8, 0);
#else
  __builtin_amdgcn_tensor_load_to_lds(g0, g1, gz, gz, 0);
#endif
}

// ------------------------------- conversions -------------------------------

__global__ void cvt_bf16_kernel(const float* __restrict__ src,
                                __bf16* __restrict__ dst, long long n) {
  long long i = (long long)blockIdx.x * blockDim.x + threadIdx.x;
  if (i < n) dst[i] = (__bf16)src[i];
}

// w: [F, E] f32 row-major  ->  wT: [E, F] bf16 (transposed)
__global__ void cvt_wT_kernel(const float* __restrict__ w,
                              __bf16* __restrict__ wT, int Fdim, int Edim) {
  long long i = (long long)blockIdx.x * blockDim.x + threadIdx.x;
  long long n = (long long)Fdim * Edim;
  if (i < n) {
    int f = (int)(i / Edim);
    int e = (int)(i % Edim);
    wT[(long long)e * Fdim + f] = (__bf16)w[i];
  }
}

// ------------------------------- WMMA GEMM ---------------------------------
// C[M,F] = A[M,K] x Bt[K,F] + bias, A bf16 [M,K], Bt bf16 [K,F] (pre-transposed weight).
// Block: 256 threads = 8 waves; block tile 128(M) x 64(F).
// K-panels of 128: B panel (128x64 bf16) staged to LDS by TDM, double buffered;
// wave w computes rows m0+16w..m0+16w+15; 4 WMMAs per 32-deep K-step.

__global__ void gemm_wmma_kernel(const __bf16* __restrict__ A,
                                 const __bf16* __restrict__ Bt,
                                 const float* __restrict__ bias,
                                 void* __restrict__ out,
                                 int M, int K, int F, int mode) {
  __shared__ __attribute__((aligned(16))) __bf16 btile[2][128 * BSTRIDE_EL];

  const int tid    = threadIdx.x;
  const int lane   = tid & 31;
  const int wslot  = tid >> 5;
  const int ftiles = F >> 6;
  const int mb = blockIdx.x / ftiles;
  const int ft = blockIdx.x % ftiles;
  const int m0 = mb << 7;                  // 128-row block strip
  const int f0 = ft << 6;                  // 64-col strip
  const int arow = m0 + (wslot << 4) + (lane & 15);
  const int koff = (lane < 16) ? 0 : 8;    // 16-bit A-fragment lane K offset

  v8f acc[4];
  v8f zero = {};
  #pragma unroll
  for (int c = 0; c < 4; ++c) acc[c] = zero;

  const int NP = K >> 7;                   // 128-deep K panels

  if (tid < 32)
    tdm_load_2d_b(Bt + f0, (unsigned int)(uintptr_t)&btile[0][0],
                  (unsigned int)F, (unsigned int)K, 64u, 128u,
                  (unsigned long long)F);

  for (int p = 0; p < NP; ++p) {
    const int cur = p & 1;
    if (tid < 32) __builtin_amdgcn_s_wait_tensorcnt(0);  // panel p landed (wave0)
    __syncthreads();                                     // publish to all waves
    if (p + 1 < NP && tid < 32)
      tdm_load_2d_b(Bt + (long long)((p + 1) << 7) * F + f0,
                    (unsigned int)(uintptr_t)&btile[1 - cur][0],
                    (unsigned int)F, (unsigned int)K, 64u, 128u,
                    (unsigned long long)F);              // overlap DMA with math

    const __bf16* abase = A + (long long)arow * K + (p << 7) + koff;
    #pragma unroll
    for (int ks = 0; ks < 4; ++ks) {
      v8bf alo = *(const v8bf*)(abase + ks * 32);
      v8bf ahi = *(const v8bf*)(abase + ks * 32 + 16);
      v16bf af;
      #pragma unroll
      for (int i = 0; i < 8; ++i) { af[i] = alo[i]; af[i + 8] = ahi[i]; }

      const __bf16* bp = &btile[cur][(ks * 32 + koff) * BSTRIDE_EL + (lane & 15)];
      #pragma unroll
      for (int c = 0; c < 4; ++c) {
        v16bf bfrag;
        #pragma unroll
        for (int i = 0; i < 8; ++i) {
          bfrag[i]     = bp[i * BSTRIDE_EL + c * 16];
          bfrag[i + 8] = bp[(16 + i) * BSTRIDE_EL + c * 16];
        }
        acc[c] = wmma_bf16(af, bfrag, acc[c]);
      }
    }
  }

  // C layout: lanes 0-15 rows 0-7 (vgpr=row), lanes 16-31 rows 8-15
  const int rbase = (lane < 16) ? 0 : 8;
  const int ncol  = lane & 15;
  #pragma unroll
  for (int c = 0; c < 4; ++c) {
    int f = f0 + c * 16 + ncol;
    float bv = bias[f];
    #pragma unroll
    for (int r = 0; r < 8; ++r) {
      int m = m0 + (wslot << 4) + rbase + r;
      float val = acc[c][r] + bv;
      if (mode == OUT_F32) {
        ((float*)out)[(long long)m * F + f] = val;
      } else if (mode == OUT_BF16_PLAIN) {
        ((__bf16*)out)[(long long)m * F + f] = (__bf16)val;
      } else if (mode == OUT_KT) {          // k -> [B,HKV,D,N]
        int b = m / N_, n = m % N_;
        int h = f / D_, d = f % D_;
        ((__bf16*)out)[((long long)(b * HKV_ + h) * D_ + d) * N_ + n] = (__bf16)val;
      } else {                              // v -> [B,HKV,N,D]
        int b = m / N_, n = m % N_;
        int h = f / D_, d = f % D_;
        ((__bf16*)out)[((long long)(b * HKV_ + h) * N_ + n) * D_ + d] = (__bf16)val;
      }
    }
  }
}

// ---------------------------- flash attention ------------------------------
// One wave per 16-query tile per (b, hq). Online softmax over 32-key blocks.
// q:[B,N,HQ,D] bf16, kT:[B,HKV,D,N] bf16, v:[B,HKV,N,D] bf16, o:[B,N,HQ,D] bf16

__global__ void flash_attn_kernel(const __bf16* __restrict__ q,
                                  const __bf16* __restrict__ kT,
                                  const __bf16* __restrict__ v,
                                  __bf16* __restrict__ o) {
  __shared__ __attribute__((aligned(32))) __bf16 lds_p[8][16 * 32];

  const int lane  = threadIdx.x & 31;
  const int wslot = threadIdx.x >> 5;
  const int wid   = blockIdx.x * (blockDim.x >> 5) + wslot;
  const int mt = wid & 127;                 // N/16 = 128 query tiles
  const int hl = wid >> 7;
  const int b  = hl / HQ_;
  const int hq = hl % HQ_;
  const int hk = hq / G_;
  const int m0 = mt << 4;

  const int koff  = (lane < 16) ? 0 : 8;
  const int rbase = (lane < 16) ? 0 : 8;
  const int ncol  = lane & 15;

  // Q A-fragments for D=64 -> two 16x32 chunks, loaded once
  const int qrow = m0 + (lane & 15);
  const __bf16* qbase = q + ((long long)(b * N_ + qrow) * HQ_ + hq) * D_;
  v16bf qf[2];
  #pragma unroll
  for (int c = 0; c < 2; ++c) {
    v8bf lo = *(const v8bf*)(qbase + c * 32 + koff);
    v8bf hi = *(const v8bf*)(qbase + c * 32 + koff + 16);
    #pragma unroll
    for (int i = 0; i < 8; ++i) { qf[c][i] = lo[i]; qf[c][i + 8] = hi[i]; }
  }

  const __bf16* kTb = kT + (long long)(b * HKV_ + hk) * D_ * N_;
  const __bf16* vb  = v  + (long long)(b * HKV_ + hk) * N_ * D_;

  v8f accd[4];
  v8f zero = {};
  #pragma unroll
  for (int c = 0; c < 4; ++c) accd[c] = zero;
  float mrun[8], lrun[8];
  #pragma unroll
  for (int r = 0; r < 8; ++r) { mrun[r] = -3.0e38f; lrun[r] = 0.0f; }

  const int nblocks = (m0 + 16 + 31) >> 5;  // causal: keys 0 .. m0+15
  for (int jb = 0; jb < nblocks; ++jb) {
    const int j0 = jb << 5;

    // ---- scores S = Q K^T (16x32 as two 16x16 f32 tiles) ----
    v8f s0 = zero, s1 = zero;
    #pragma unroll
    for (int c = 0; c < 2; ++c) {
      const __bf16* kp = kTb + (long long)(c * 32 + koff) * N_ + j0 + (lane & 15);
      v16bf b0, b1;
      #pragma unroll
      for (int i = 0; i < 8; ++i) {
        b0[i] = kp[(long long)i * N_];      b0[i + 8] = kp[(long long)(16 + i) * N_];
        b1[i] = kp[(long long)i * N_ + 16]; b1[i + 8] = kp[(long long)(16 + i) * N_ + 16];
      }
      s0 = wmma_bf16(qf[c], b0, s0);
      s1 = wmma_bf16(qf[c], b1, s1);
    }

    // ---- scale + causal mask ----
    const bool partial = (j0 + 31 > m0);
    float p0[8], p1[8];
    #pragma unroll
    for (int r = 0; r < 8; ++r) {
      int m = m0 + rbase + r;
      float x0 = s0[r] * SCALE_;
      float x1 = s1[r] * SCALE_;
      if (partial) {
        if (j0 + ncol > m)      x0 = -1.0e30f;
        if (j0 + 16 + ncol > m) x1 = -1.0e30f;
      }
      p0[r] = x0; p1[r] = x1;
    }

    // ---- online softmax (row reductions across 16-lane halves) ----
    #pragma unroll
    for (int r = 0; r < 8; ++r) {
      float mx = fmaxf(p0[r], p1[r]);
      #pragma unroll
      for (int off = 1; off < 16; off <<= 1)
        mx = fmaxf(mx, __shfl_xor(mx, off, 32));
      float mnew = fmaxf(mrun[r], mx);
      float e0 = __expf(p0[r] - mnew);
      float e1 = __expf(p1[r] - mnew);
      float rs = e0 + e1;
      #pragma unroll
      for (int off = 1; off < 16; off <<= 1)
        rs += __shfl_xor(rs, off, 32);
      float alpha = __expf(mrun[r] - mnew);
      lrun[r] = lrun[r] * alpha + rs;
      mrun[r] = mnew;
      #pragma unroll
      for (int c = 0; c < 4; ++c) accd[c][r] *= alpha;
      p0[r] = e0; p1[r] = e1;
    }

    // ---- C-layout -> A-fragment layout for P via per-wave LDS tile ----
    __bf16* lp = &lds_p[wslot][0];
    #pragma unroll
    for (int r = 0; r < 8; ++r) {
      lp[(rbase + r) * 32 + ncol]      = (__bf16)p0[r];
      lp[(rbase + r) * 32 + 16 + ncol] = (__bf16)p1[r];
    }
    asm volatile("s_wait_dscnt 0" ::: "memory");  // stores visible to all lanes

    v16bf pf;
    {
      const __bf16* pr = lp + (lane & 15) * 32 + koff;
      v8bf lo = *(const v8bf*)(pr);
      v8bf hi = *(const v8bf*)(pr + 16);
      #pragma unroll
      for (int i = 0; i < 8; ++i) { pf[i] = lo[i]; pf[i + 8] = hi[i]; }
    }

    // ---- O += P V  (4 d-chunks of 16) ----
    const __bf16* vp = vb + (long long)(j0 + koff) * D_ + (lane & 15);
    #pragma unroll
    for (int c = 0; c < 4; ++c) {
      v16bf bfv;
      #pragma unroll
      for (int i = 0; i < 8; ++i) {
        bfv[i]     = vp[i * D_ + c * 16];
        bfv[i + 8] = vp[(16 + i) * D_ + c * 16];
      }
      accd[c] = wmma_bf16(pf, bfv, accd[c]);
    }
  }

  // ---- epilogue: O / l, write bf16 [B,N,HQ,D] ----
  __bf16* ob = o + ((long long)(b * N_ + m0) * HQ_ + hq) * D_;
  #pragma unroll
  for (int c = 0; c < 4; ++c) {
    #pragma unroll
    for (int r = 0; r < 8; ++r) {
      int m = rbase + r;
      float val = accd[c][r] / lrun[r];
      ob[(long long)m * (HQ_ * D_) + c * 16 + ncol] = (__bf16)val;
    }
  }
}

// --------------------------------- launch ----------------------------------

extern "C" void kernel_launch(void* const* d_in, const int* in_sizes, int n_in,
                              void* d_out, int out_size, void* d_ws, size_t ws_size,
                              hipStream_t stream) {
  const float* x  = (const float*)d_in[0];
  const float* Wq = (const float*)d_in[1];
  const float* bq = (const float*)d_in[2];
  const float* Wk = (const float*)d_in[3];
  const float* bk = (const float*)d_in[4];
  const float* Wv = (const float*)d_in[5];
  const float* bv = (const float*)d_in[6];
  const float* Wo = (const float*)d_in[7];
  const float* bo = (const float*)d_in[8];
  float* out = (float*)d_out;

  char* ws = (char*)d_ws;
  size_t off = 0;
  auto salloc = [&](size_t bytes) -> void* {
    void* p = ws + off;
    off = (off + bytes + 255) & ~(size_t)255;
    return p;
  };
  __bf16* xb   = (__bf16*)salloc((size_t)B_ * N_ * E_ * 2);
  __bf16* WqT  = (__bf16*)salloc((size_t)E_ * HQ_ * D_ * 2);
  __bf16* WkT  = (__bf16*)salloc((size_t)E_ * HKV_ * D_ * 2);
  __bf16* WvT  = (__bf16*)salloc((size_t)E_ * HKV_ * D_ * 2);
  __bf16* WoT  = (__bf16*)salloc((size_t)E_ * E_ * 2);
  __bf16* qb   = (__bf16*)salloc((size_t)B_ * N_ * HQ_ * D_ * 2);
  __bf16* kTb  = (__bf16*)salloc((size_t)B_ * HKV_ * D_ * N_ * 2);
  __bf16* vbuf = (__bf16*)salloc((size_t)B_ * HKV_ * N_ * D_ * 2);
  __bf16* obuf = (__bf16*)salloc((size_t)B_ * N_ * E_ * 2);

  // 1) casts (x) and transposed casts (weights)
  long long nx = (long long)B_ * N_ * E_;
  cvt_bf16_kernel<<<(int)((nx + 255) / 256), 256, 0, stream>>>(x, xb, nx);
  cvt_wT_kernel<<<(HQ_ * D_ * E_) / 256, 256, 0, stream>>>(Wq, WqT, HQ_ * D_, E_);
  cvt_wT_kernel<<<(HKV_ * D_ * E_) / 256, 256, 0, stream>>>(Wk, WkT, HKV_ * D_, E_);
  cvt_wT_kernel<<<(HKV_ * D_ * E_) / 256, 256, 0, stream>>>(Wv, WvT, HKV_ * D_, E_);
  cvt_wT_kernel<<<(E_ * E_) / 256, 256, 0, stream>>>(Wo, WoT, E_, E_);

  const int M = B_ * N_;  // 4096 token rows

  // 2) projections (TDM-staged WMMA GEMMs)
  { int F = HQ_ * D_;  int blocks = (M / 128) * (F / 64);
    gemm_wmma_kernel<<<blocks, 256, 0, stream>>>(xb, WqT, bq, qb, M, E_, F, OUT_BF16_PLAIN); }
  { int F = HKV_ * D_; int blocks = (M / 128) * (F / 64);
    gemm_wmma_kernel<<<blocks, 256, 0, stream>>>(xb, WkT, bk, kTb, M, E_, F, OUT_KT); }
  { int F = HKV_ * D_; int blocks = (M / 128) * (F / 64);
    gemm_wmma_kernel<<<blocks, 256, 0, stream>>>(xb, WvT, bv, vbuf, M, E_, F, OUT_V); }

  // 3) causal flash attention (one wave per 16-query tile per head)
  { int tiles = B_ * HQ_ * (N_ / 16);  // 8192 waves
    flash_attn_kernel<<<tiles / 8, 256, 0, stream>>>(qb, kTb, vbuf, obuf); }

  // 4) output projection, f32 + bias directly into d_out
  { int F = E_; int blocks = (M / 128) * (F / 64);
    gemm_wmma_kernel<<<blocks, 256, 0, stream>>>(obuf, WoT, bo, out, M, E_, F, OUT_F32); }
}
